// InfiniAttention_45612552684059
// MI455X (gfx1250) — matrix operand order repaired and checked by reading, not compiled, not run on hardware
//
#include <hip/hip_runtime.h>
#include <hip/hip_bf16.h>

typedef __attribute__((ext_vector_type(16))) __bf16 bf16x16;
typedef __attribute__((ext_vector_type(8)))  float  f32x8;
typedef __attribute__((ext_vector_type(4)))  unsigned int v4u;
typedef __attribute__((ext_vector_type(8)))  int v8i;
typedef __attribute__((ext_vector_type(4)))  int v4i;

typedef const __attribute__((address_space(3))) __bf16* lds_cbf16p;

#define SEQ    6400
#define BATCH  2
#define DMODEL 2048
#define NHEADS 16
#define DHEAD  128
#define NSEG   50
#define SEGLEN 128
#define E3     6144   /* 3*DMODEL */

#ifndef __has_builtin
#define __has_builtin(x) 0
#endif
#if __has_builtin(__builtin_amdgcn_tensor_load_to_lds) && __has_builtin(__builtin_amdgcn_s_wait_tensorcnt)
#define HAVE_TDM 1
#else
#define HAVE_TDM 0
#endif

static __device__ __forceinline__ f32x8 wmma_bf16(bf16x16 a, bf16x16 b, f32x8 c) {
  return __builtin_amdgcn_wmma_f32_16x16x32_bf16(false, a, false, b, (short)0, c, false, false);
}

/* ---- WMMA fragment loaders (contiguous k order: k = 16*(lane/16) + s); PT may be generic or AS3 pointer ---- */
template <typename PT>
static __device__ __forceinline__ bf16x16 load_a(PT base, int ld, int lane) {
  int row = lane & 15;
  int kh  = (lane >> 4) << 4;
  PT p = base + row * ld + kh;
  bf16x16 a;
#pragma unroll
  for (int i = 0; i < 8; ++i) a[i] = p[i];
#pragma unroll
  for (int i = 0; i < 8; ++i) a[8 + i] = p[8 + i];
  return a;
}
template <typename PT>
static __device__ __forceinline__ bf16x16 load_at(PT base, int ld, int lane) {
  int row = lane & 15;
  int kh  = (lane >> 4) << 4;
  bf16x16 a;
#pragma unroll
  for (int i = 0; i < 8; ++i) a[i] = base[(kh + i) * ld + row];
#pragma unroll
  for (int i = 0; i < 8; ++i) a[8 + i] = base[(kh + 8 + i) * ld + row];
  return a;
}
template <typename PT>
static __device__ __forceinline__ bf16x16 load_b(PT base, int ld, int lane) {
  int col  = lane & 15;
  int kOff = (lane >> 4) << 4;
  PT p = base + kOff * ld + col;
  bf16x16 b;
#pragma unroll
  for (int i = 0; i < 16; ++i) b[i] = p[i * ld];
  return b;
}
static __device__ __forceinline__ bf16x16 load_b_f32(const float* base, int ld, int lane) {
  int col  = lane & 15;
  int kOff = (lane >> 4) << 4;
  const float* p = base + kOff * ld + col;
  bf16x16 b;
#pragma unroll
  for (int i = 0; i < 16; ++i) b[i] = (__bf16)p[i * ld];
  return b;
}
template <typename PT>
static __device__ __forceinline__ bf16x16 load_bt(PT base, int ld, int lane) {
  int col  = lane & 15;
  int kOff = (lane >> 4) << 4;
  PT p = base + col * ld + kOff;
  bf16x16 b;
#pragma unroll
  for (int i = 0; i < 16; ++i) b[i] = p[i];
  return b;
}

static __device__ __forceinline__ unsigned lds_addr_of(const void* p) {
  return (unsigned)(unsigned long long)(const __attribute__((address_space(3))) char*)p;
}

/* Hide the LDS offset from the optimizer (TDM writes LDS invisibly), but keep the
   pointer in addrspace(3) so reads still lower to ds_load_*. */
static __device__ __forceinline__ lds_cbf16p launder_lds(const void* p) {
  unsigned off = lds_addr_of(p);
  asm volatile("" : "+v"(off));
  return (lds_cbf16p)(unsigned long long)off;
}

#if HAVE_TDM
/* 2D tile load via Tensor Data Mover (D# per CDNA5 ISA).
   data_size = 2 bytes. Optional LDS row padding: after every (pad-interval) DWORDs
   stored, skip pad_amount DWORDs. */
static __device__ __forceinline__ void tdm_load_2d(
    unsigned lds_byte_addr, const void* gaddr,
    unsigned tensor_d0, unsigned tensor_d1,
    unsigned tile_d0, unsigned tile_d1,
    unsigned long long stride0_elems,
    unsigned pad_interval_code, unsigned pad_amount_code, bool pad_en)
{
  unsigned long long ga = (unsigned long long)gaddr;
  v4u g0;
  g0[0] = 1u;                                             // count=1, user desc
  g0[1] = lds_byte_addr;                                  // lds_addr
  g0[2] = (unsigned)(ga & 0xffffffffu);                   // global_addr[31:0]
  g0[3] = (unsigned)((ga >> 32) & 0x01ffffffu) | (2u << 30); // addr[56:32] | type=2
  unsigned w0 = (1u << 16);                               // data_size=1 (2B), wg_mask=0
  if (pad_en) w0 |= (1u << 20) | (pad_interval_code << 22) | (pad_amount_code << 25);
  v8i g1;
  g1[0] = (int)w0;
  g1[1] = (int)((tensor_d0 & 0xffffu) << 16);             // tensor_dim0[15:0] in [31:16]
  g1[2] = (int)(((tensor_d0 >> 16) & 0xffffu) | ((tensor_d1 & 0xffffu) << 16));
  g1[3] = (int)(((tensor_d1 >> 16) & 0xffffu) | ((tile_d0 & 0xffffu) << 16));
  g1[4] = (int)(tile_d1 & 0xffffu);                       // tile_dim1 | tile_dim2=0
  g1[5] = (int)(unsigned)(stride0_elems & 0xffffffffu);   // tensor_dim0_stride[31:0]
  g1[6] = (int)(unsigned)((stride0_elems >> 32) & 0xffffu); // stride[47:32] | d1_stride lo=0
  g1[7] = 0;
  v4i g2 = {0, 0, 0, 0};
  v4i g3 = {0, 0, 0, 0};
  v8i g4 = {0, 0, 0, 0, 0, 0, 0, 0};
  // amdgpu-toolchain (clang-23) 6-arg form: (g0, g1, g2, g3, <v8i>, cpol)
  __builtin_amdgcn_tensor_load_to_lds(g0, g1, g2, g3, g4, 0);
}
#endif

/* ---------------- fp32 -> bf16 convert ---------------- */
__global__ void cvt_f32_bf16(const float* __restrict__ in, __bf16* __restrict__ out, long long n) {
  long long i = (long long)blockIdx.x * blockDim.x + threadIdx.x;
  long long stride = (long long)gridDim.x * blockDim.x;
  for (; i < n; i += stride) out[i] = (__bf16)in[i];
}

/* ---------------- big tiled GEMM: C[MxN] = A[MxK]*B[KxN], bf16 in, OutT out -------------
   128x128 block tile, 8 waves as 4x2 (each 32x64 = 2x4 WMMA tiles), K-step 32.
   TDM double-buffered staging when available. */
template <typename OutT>
__global__ __launch_bounds__(256) void gemm_bf16_kernel(
    const __bf16* __restrict__ A, const __bf16* __restrict__ B,
    OutT* __restrict__ C, int M, int N, int K)
{
  __shared__ __bf16 As[2][128 * 40];   // ld 40 (pad 8)
  __shared__ __bf16 Bs[2][32 * 136];   // ld 136 (pad 8)

  const int tid  = threadIdx.x;
  const int lane = tid & 31, wave = tid >> 5;
  const int wr = wave >> 1, wc = wave & 1;
  const int bm = blockIdx.y * 128, bn = blockIdx.x * 128;
  const int nIdx = lane & 15, mBase = (lane >> 4) << 3;

  f32x8 acc[2][4] = {};
  const int nsteps = K >> 5;

#if HAVE_TDM
  // stage k0: A tile 128x32 (row=16 DW + 4 DW pad -> ld 40), B tile 32x128 (64 DW + 4 DW -> ld 136)
  if (wave == 0) {
    tdm_load_2d(lds_addr_of(&As[0][0]), A + (size_t)bm * K, K, M, 32, 128, (unsigned long long)K, 3u, 3u, true);
    tdm_load_2d(lds_addr_of(&Bs[0][0]), B + (size_t)bn, N, K, 128, 32, (unsigned long long)N, 5u, 3u, true);
    if (nsteps > 1) {
      tdm_load_2d(lds_addr_of(&As[1][0]), A + (size_t)bm * K + 32, K, M, 32, 128, (unsigned long long)K, 3u, 3u, true);
      tdm_load_2d(lds_addr_of(&Bs[1][0]), B + (size_t)32 * N + bn, N, K, 128, 32, (unsigned long long)N, 5u, 3u, true);
    }
  }
  for (int it = 0; it < nsteps; ++it) {
    const int buf = it & 1;
    if (wave == 0) {
      if (it + 1 < nsteps) __builtin_amdgcn_s_wait_tensorcnt(2);
      else                 __builtin_amdgcn_s_wait_tensorcnt(0);
    }
    __syncthreads();
    lds_cbf16p as = launder_lds(&As[buf][0]);
    lds_cbf16p bs = launder_lds(&Bs[buf][0]);
    bf16x16 af0 = load_a(as + (wr * 32) * 40,      40, lane);
    bf16x16 af1 = load_a(as + (wr * 32 + 16) * 40, 40, lane);
#pragma unroll
    for (int ni = 0; ni < 4; ++ni) {
      bf16x16 bf = load_b(bs + wc * 64 + ni * 16, 136, lane);
      acc[0][ni] = wmma_bf16(af0, bf, acc[0][ni]);
      acc[1][ni] = wmma_bf16(af1, bf, acc[1][ni]);
    }
    __syncthreads();
    const int k2 = (it + 2) * 32;
    if (wave == 0 && k2 < K) {
      tdm_load_2d(lds_addr_of(&As[buf][0]), A + (size_t)bm * K + k2, K, M, 32, 128, (unsigned long long)K, 3u, 3u, true);
      tdm_load_2d(lds_addr_of(&Bs[buf][0]), B + (size_t)k2 * N + bn, N, K, 128, 32, (unsigned long long)N, 5u, 3u, true);
    }
  }
#else
  const int ar = tid >> 1, ac = (tid & 1) << 4;
  const int br = tid >> 3, bc = (tid & 7) << 4;
  for (int it = 0; it < nsteps; ++it) {
    const int k0 = it * 32;
    __syncthreads();
    {
      const uint4* s4 = (const uint4*)(A + (size_t)(bm + ar) * K + k0 + ac);
      uint4 v0 = s4[0], v1 = s4[1];
      *(uint4*)&As[0][ar * 40 + ac]     = v0;
      *(uint4*)&As[0][ar * 40 + ac + 8] = v1;
    }
    {
      const uint4* s4 = (const uint4*)(B + (size_t)(k0 + br) * N + bn + bc);
      uint4 v0 = s4[0], v1 = s4[1];
      *(uint4*)&Bs[0][br * 136 + bc]     = v0;
      *(uint4*)&Bs[0][br * 136 + bc + 8] = v1;
    }
    __syncthreads();
    bf16x16 af0 = load_a((const __bf16*)&As[0][(wr * 32) * 40],      40, lane);
    bf16x16 af1 = load_a((const __bf16*)&As[0][(wr * 32 + 16) * 40], 40, lane);
#pragma unroll
    for (int ni = 0; ni < 4; ++ni) {
      bf16x16 bf = load_b((const __bf16*)&Bs[0][wc * 64 + ni * 16], 136, lane);
      acc[0][ni] = wmma_bf16(af0, bf, acc[0][ni]);
      acc[1][ni] = wmma_bf16(af1, bf, acc[1][ni]);
    }
  }
#endif

#pragma unroll
  for (int mi = 0; mi < 2; ++mi)
#pragma unroll
    for (int ni = 0; ni < 4; ++ni)
#pragma unroll
      for (int i = 0; i < 8; ++i) {
        int gr = bm + wr * 32 + mi * 16 + mBase + i;
        int gc = bn + wc * 64 + ni * 16 + nIdx;
        C[(size_t)gr * N + gc] = (OutT)acc[mi][ni][i];
      }
}

/* ---------------- local causal attention, one WG per (seg,b,h) ---------------- */
__global__ __launch_bounds__(256) void local_attn_kernel(
    const __bf16* __restrict__ qkv, const unsigned char* __restrict__ seqmask,
    __bf16* __restrict__ localout)
{
  __shared__ __bf16 Qs[SEGLEN * 136];  // reused to hold softmax P
  __shared__ __bf16 Ks[SEGLEN * 136];
  __shared__ __bf16 Vs[SEGLEN * 136];
  __shared__ unsigned char km[SEGLEN];

  const int h = blockIdx.x, b = blockIdx.y, n = blockIdx.z;
  const int tid = threadIdx.x;
  const int lane = tid & 31, wave = tid >> 5;
  const int m0 = wave * 16;
  const int nIdx = lane & 15, mBase = (lane >> 4) << 3;

#if HAVE_TDM
  // Q/K/V tiles: 128 rows x 128 cols, row stride 2*E3 elements (batch interleave).
  // Rows are 64 DWORDs; pad +4 DWORDs -> LDS ld 136.
  if (wave == 0) {
    const __bf16* base = qkv + (size_t)(n * SEGLEN * BATCH + b) * E3 + h * DHEAD;
    tdm_load_2d(lds_addr_of(Qs), base,              E3, (unsigned)(SEGLEN * BATCH), DHEAD, SEGLEN,
                (unsigned long long)(2 * E3), 5u, 3u, true);
    tdm_load_2d(lds_addr_of(Ks), base + DMODEL,     E3, (unsigned)(SEGLEN * BATCH), DHEAD, SEGLEN,
                (unsigned long long)(2 * E3), 5u, 3u, true);
    tdm_load_2d(lds_addr_of(Vs), base + 2 * DMODEL, E3, (unsigned)(SEGLEN * BATCH), DHEAD, SEGLEN,
                (unsigned long long)(2 * E3), 5u, 3u, true);
    __builtin_amdgcn_s_wait_tensorcnt(0);
  }
#else
  for (int idx = tid; idx < SEGLEN * 16; idx += 256) {
    int s  = idx >> 4;
    int c8 = (idx & 15) << 3;
    size_t row = (size_t)((n * SEGLEN + s) * BATCH + b) * E3 + h * DHEAD + c8;
    *(uint4*)&Qs[s * 136 + c8] = *(const uint4*)(qkv + row);
    *(uint4*)&Ks[s * 136 + c8] = *(const uint4*)(qkv + row + DMODEL);
    *(uint4*)&Vs[s * 136 + c8] = *(const uint4*)(qkv + row + 2 * DMODEL);
  }
#endif
  if (tid < SEGLEN) km[tid] = seqmask[b * SEQ + n * SEGLEN + tid];
  __syncthreads();

  lds_cbf16p qs = launder_lds(Qs);
  lds_cbf16p ks = launder_lds(Ks);
  lds_cbf16p vs = launder_lds(Vs);

  // scores = Q * K^T (wave owns a 16-row stripe x all 128 cols)
  f32x8 acc[8] = {};
#pragma unroll
  for (int k0 = 0; k0 < DHEAD; k0 += 32) {
    bf16x16 af = load_a(qs + m0 * 136 + k0, 136, lane);
#pragma unroll
    for (int ni = 0; ni < 8; ++ni) {
      bf16x16 bf = load_bt(ks + (ni * 16) * 136 + k0, 136, lane);
      acc[ni] = wmma_bf16(af, bf, acc[ni]);
    }
  }

  const float scale = 0.088388347648318447f;  // 1/sqrt(128)
#pragma unroll
  for (int ni = 0; ni < 8; ++ni)
#pragma unroll
    for (int i = 0; i < 8; ++i) {
      int srow = m0 + mBase + i;
      int scol = ni * 16 + nIdx;
      bool keep = (scol <= srow) && (km[scol] != 0);
      acc[ni][i] = keep ? acc[ni][i] * scale : -1.0e9f;
    }

  // softmax: each row lives across one half-wave (16 lanes) x 8 tiles
#pragma unroll
  for (int i = 0; i < 8; ++i) {
    float mx = acc[0][i];
#pragma unroll
    for (int ni = 1; ni < 8; ++ni) mx = fmaxf(mx, acc[ni][i]);
#pragma unroll
    for (int d = 1; d < 16; d <<= 1) mx = fmaxf(mx, __shfl_xor(mx, d, 16));
    float sum = 0.f;
#pragma unroll
    for (int ni = 0; ni < 8; ++ni) { float e = __expf(acc[ni][i] - mx); acc[ni][i] = e; sum += e; }
#pragma unroll
    for (int d = 1; d < 16; d <<= 1) sum += __shfl_xor(sum, d, 16);
    float inv = 1.f / sum;
    int srow = m0 + mBase + i;
#pragma unroll
    for (int ni = 0; ni < 8; ++ni)
      Qs[srow * 136 + ni * 16 + nIdx] = (__bf16)(acc[ni][i] * inv);   // own stripe only
  }
  __syncthreads();

  // out = P * V
  f32x8 o[8] = {};
#pragma unroll
  for (int k0 = 0; k0 < SEGLEN; k0 += 32) {
    bf16x16 af = load_a(qs + m0 * 136 + k0, 136, lane);
#pragma unroll
    for (int ni = 0; ni < 8; ++ni) {
      bf16x16 bf = load_b(vs + k0 * 136 + ni * 16, 136, lane);
      o[ni] = wmma_bf16(af, bf, o[ni]);
    }
  }
#pragma unroll
  for (int ni = 0; ni < 8; ++ni)
#pragma unroll
    for (int i = 0; i < 8; ++i) {
      int srow = m0 + mBase + i;
      size_t gr = (size_t)((n * SEGLEN + srow) * BATCH + b);
      localout[gr * DMODEL + h * DHEAD + ni * 16 + nIdx] = (__bf16)o[ni][i];
    }
}

/* ---------------- compressive-memory scan, one WG per (b,h); mem fp32 in LDS ---------------- */
__global__ __launch_bounds__(256) void scan_kernel(
    const __bf16* __restrict__ qkv, const __bf16* __restrict__ localb,
    const float* __restrict__ bal, __bf16* __restrict__ attnb)
{
  __shared__ float  Mem[DHEAD * DHEAD];   // [d][e], fp32 master copy
  __shared__ __bf16 SQ[SEGLEN * 136];     // phi(q); reused for new_v
  __shared__ __bf16 SK[SEGLEN * 136];     // phi(k)
  __shared__ float  Norm[DHEAD], DenQ[SEGLEN], DenK[SEGLEN], SumK[DHEAD];

  const int h = blockIdx.x, b = blockIdx.y;
  const int tid = threadIdx.x;
  const int lane = tid & 31, wave = tid >> 5;
  const int m0 = wave * 16;
  const int nIdx = lane & 15, mBase = (lane >> 4) << 3;

  const float g = 1.f / (1.f + __expf(-bal[h]));

  for (int i = tid; i < DHEAD * DHEAD; i += 256) Mem[i] = 0.f;
  if (tid < DHEAD) Norm[tid] = 0.f;
  __syncthreads();

  for (int n = 0; n < NSEG; ++n) {
    // load phi(q)=elu(q)+1, phi(k)=elu(k)+1
    for (int idx = tid; idx < SEGLEN * 16; idx += 256) {
      int s  = idx >> 4;
      int c8 = (idx & 15) << 3;
      size_t rowoff = (size_t)((n * SEGLEN + s) * BATCH + b) * E3 + h * DHEAD + c8;
      uint4 qv = *(const uint4*)(qkv + rowoff);
      uint4 kv = *(const uint4*)(qkv + rowoff + DMODEL);
      const __bf16* qh = (const __bf16*)&qv;
      const __bf16* kp = (const __bf16*)&kv;
#pragma unroll
      for (int j = 0; j < 8; ++j) {
        float qf = (float)qh[j], kf = (float)kp[j];
        SQ[s * 136 + c8 + j] = (__bf16)(qf > 0.f ? qf + 1.f : __expf(qf));
        SK[s * 136 + c8 + j] = (__bf16)(kf > 0.f ? kf + 1.f : __expf(kf));
      }
    }
    __syncthreads();

    if (tid < SEGLEN) {
      int r = tid;
      float dq = 0.f, dk = 0.f, sk = 0.f;
      for (int d = 0; d < DHEAD; ++d) {
        dq += (float)SQ[r * 136 + d] * Norm[d];
        dk += (float)SK[r * 136 + d] * Norm[d];
        sk += (float)SK[d * 136 + r];          // column sum of phi(k)
      }
      DenQ[r] = dq; DenK[r] = dk; SumK[r] = sk;
    }
    __syncthreads();

    // retr = SQ*Mem ; prev_v = SK*Mem   (Mem fragments converted fp32->bf16 on load)
    f32x8 accR[8] = {}, accP[8] = {};
#pragma unroll
    for (int k0 = 0; k0 < DHEAD; k0 += 32) {
      bf16x16 aq = load_a((const __bf16*)&SQ[m0 * 136 + k0], 136, lane);
      bf16x16 ak = load_a((const __bf16*)&SK[m0 * 136 + k0], 136, lane);
#pragma unroll
      for (int ni = 0; ni < 8; ++ni) {
        bf16x16 bm = load_b_f32(&Mem[k0 * DHEAD + ni * 16], DHEAD, lane);
        accR[ni] = wmma_bf16(aq, bm, accR[ni]);
        accP[ni] = wmma_bf16(ak, bm, accP[ni]);
      }
    }

    // gate+combine with local attention; compute new_v into SQ (own stripe only)
#pragma unroll
    for (int ni = 0; ni < 8; ++ni)
#pragma unroll
      for (int i = 0; i < 8; ++i) {
        int srow = m0 + mBase + i;
        int scol = ni * 16 + nIdx;
        size_t grow = (size_t)((n * SEGLEN + srow) * BATCH + b);
        float retr  = (n > 0) ? accR[ni][i] / DenQ[srow] : 0.f;
        float loc   = (float)localb[grow * DMODEL + h * DHEAD + scol];
        attnb[grow * DMODEL + h * DHEAD + scol] = (__bf16)(g * retr + (1.f - g) * loc);
        float prevv = (n > 0) ? accP[ni][i] / DenK[srow] : 0.f;
        float vv    = (float)qkv[grow * E3 + 2 * DMODEL + h * DHEAD + scol];
        SQ[srow * 136 + scol] = (__bf16)(vv - prevv);
      }
    __syncthreads();

    // Mem += SK^T * new_v
    f32x8 accM[8] = {};
#pragma unroll
    for (int k0 = 0; k0 < SEGLEN; k0 += 32) {
      bf16x16 at = load_at((const __bf16*)&SK[k0 * 136 + m0], 136, lane);
#pragma unroll
      for (int ni = 0; ni < 8; ++ni) {
        bf16x16 bv = load_b((const __bf16*)&SQ[k0 * 136 + ni * 16], 136, lane);
        accM[ni] = wmma_bf16(at, bv, accM[ni]);
      }
    }
#pragma unroll
    for (int ni = 0; ni < 8; ++ni)
#pragma unroll
      for (int i = 0; i < 8; ++i)
        Mem[(m0 + mBase + i) * DHEAD + ni * 16 + nIdx] += accM[ni][i];
    __syncthreads();
    if (tid < DHEAD) Norm[tid] += SumK[tid];
    __syncthreads();
  }
}

/* ---------------- launch ---------------- */
extern "C" void kernel_launch(void* const* d_in, const int* in_sizes, int n_in,
                              void* d_out, int out_size, void* d_ws, size_t ws_size,
                              hipStream_t stream) {
  const float*         x    = (const float*)d_in[0];
  const unsigned char* mask = (const unsigned char*)d_in[1];
  const float*         wqkv = (const float*)d_in[2];
  const float*         wo   = (const float*)d_in[3];
  const float*         bal  = (const float*)d_in[4];
  float* out = (float*)d_out;

  char* ws = (char*)d_ws;
  size_t off = 0;
  auto carve = [&](size_t bytes) -> void* {
    void* p = ws + off;
    off += (bytes + 255) & ~(size_t)255;
    return p;
  };
  const size_t ROWS = (size_t)SEQ * BATCH;            // 12800
  __bf16* xb     = (__bf16*)carve(ROWS * DMODEL * 2);
  __bf16* wqkvb  = (__bf16*)carve((size_t)DMODEL * E3 * 2);
  __bf16* wob    = (__bf16*)carve((size_t)DMODEL * DMODEL * 2);
  __bf16* qkvb   = (__bf16*)carve(ROWS * E3 * 2);
  __bf16* localb = (__bf16*)carve(ROWS * DMODEL * 2);
  __bf16* attnb  = (__bf16*)carve(ROWS * DMODEL * 2);

  cvt_f32_bf16<<<2048, 256, 0, stream>>>(x,    xb,    (long long)ROWS * DMODEL);
  cvt_f32_bf16<<<2048, 256, 0, stream>>>(wqkv, wqkvb, (long long)DMODEL * E3);
  cvt_f32_bf16<<<1024, 256, 0, stream>>>(wo,   wob,   (long long)DMODEL * DMODEL);

  // qkv = x @ w_qkv : [12800 x 6144]
  gemm_bf16_kernel<__bf16><<<dim3(E3 / 128, ROWS / 128), 256, 0, stream>>>(
      xb, wqkvb, qkvb, (int)ROWS, E3, DMODEL);

  // per-segment causal attention
  local_attn_kernel<<<dim3(NHEADS, BATCH, NSEG), 256, 0, stream>>>(qkvb, mask, localb);

  // sequential compressive-memory scan + gating -> combined attn (bf16)
  scan_kernel<<<dim3(NHEADS, BATCH), 256, 0, stream>>>(qkvb, localb, bal, attnb);

  // out = attn @ w_o : [12800 x 2048], fp32
  gemm_bf16_kernel<float><<<dim3(DMODEL / 128, ROWS / 128), 256, 0, stream>>>(
      attnb, wob, out, (int)ROWS, DMODEL, DMODEL);
}
